// Quantization_25812753449605
// MI455X (gfx1250) — matrix operand order repaired and checked by reading, not compiled, not run on hardware
//
#include <hip/hip_runtime.h>

// soft-codebook quantization, t-Student kernel, STE output.
// Log2-domain weights (fp64 dynamic range emulated in fp32), EPS blend via
// max-normalization, and the reference's weights@[cb|1] matmul done with
// v_wmma_f32_16x16x32_f16 (one WMMA per 16-element tile, wave32).

typedef _Float16 v16h __attribute__((ext_vector_type(16)));
typedef float    v8f  __attribute__((ext_vector_type(8)));

#define G2_OVER_V 12.5f        // GAMMA^2 / V = 625/50
#define NEG_EXP   25.5f        // (V+1)/2
#define L_EPS   (-239.178823f) // log2(1e-72)

__device__ __forceinline__ float fast_exp2(float a) { return __builtin_amdgcn_exp2f(a); } // v_exp_f32
__device__ __forceinline__ float fast_log2(float a) { return __builtin_amdgcn_logf(a); }  // v_log_f32
__device__ __forceinline__ float fast_rcp(float a)  { return __builtin_amdgcn_rcpf(a); }  // v_rcp_f32

__device__ __forceinline__ float hardQ(float x) {
  // nearest codeword in {-7..8}; ties -> lower index (matches argmax-first)
  return fminf(fmaxf(ceilf(x - 0.5f), -7.0f), 8.0f);
}

__global__ __launch_bounds__(256) void quant_ts_wmma(const float* __restrict__ x,
                                                     float* __restrict__ out,
                                                     int nTiles, int N) {
  const int lane    = threadIdx.x & 31;
  const int r       = lane & 15;            // row (element) within 16-elem tile
  const int kBase   = (lane & 16) ? 8 : 0;  // this lane's codeword sub-range
  const int waveId  = (blockIdx.x * blockDim.x + threadIdx.x) >> 5;
  const int nWaves  = (gridDim.x * blockDim.x) >> 5;

  // B matrix (f16 32x16): col 0 = codebook, col 1 = ones, rest 0; rows 16..31 = 0.
  // B layout matches A: lane n (n<16) elems 0..7 -> rows 0..7, lane n+16 elems 0..7 -> rows 8..15,
  // column = lane % 16; elems 8..15 (rows 16..31) stay zero.
  v16h bm = {};
  {
    if (r == 0) {
#pragma unroll
      for (int k = 0; k < 8; ++k) bm[k] = (_Float16)(float)(kBase + k - 7);
    } else if (r == 1) {
#pragma unroll
      for (int k = 0; k < 8; ++k) bm[k] = (_Float16)1.0f;
    }
  }

  for (int tile = waveId; tile < nTiles; tile += nWaves) {   // wave-uniform loop
    const float xv = x[tile * 16 + r];   // lanes n and n+16 both hold element n

    // log2-domain t-Student weights for this lane's 8 codewords
    float L[8], m = -1e30f;
#pragma unroll
    for (int j = 0; j < 8; ++j) {
      float d = xv - (float)(kBase + j - 7);
      float t = fmaf(d * d, G2_OVER_V, 1.0f);
      L[j] = -NEG_EXP * fast_log2(t);
      m = fmaxf(m, L[j]);
    }
    m = fmaxf(m, __shfl_xor(m, 16, 32));     // max over all 16 codewords
    const float m2 = fmaxf(m, L_EPS);        // keep EPS blend in-range
    const float eT = fast_exp2(L_EPS - m2);  // normalized EPS term

    // A matrix (f16 16x32): lane n -> row n K=0..7, lane n+16 -> row n K=8..15,
    // K=16..31 (vector elems 8..15) stay zero.
    v16h am = {};
#pragma unroll
    for (int j = 0; j < 8; ++j)
      am[j] = (_Float16)(fast_exp2(L[j] - m2) + eT);

    v8f c = {};
    c = __builtin_amdgcn_wmma_f32_16x16x32_f16(false, am, false, bm,
                                               (short)0, c, false, false);
    // D layout: num (col 0) -> lane 0 (M=0..7 in c[0..7]) and lane 16 (M=8..15);
    //           den (col 1) -> lanes 1 / 17.
    float den[8], xs[8];
    const int srcBase = (lane & 16) ? 8 : 0;
#pragma unroll
    for (int g = 0; g < 8; ++g) {
      den[g] = __shfl_xor(c[g], 1, 32);      // pair num-lane with den-lane
      xs[g]  = __shfl(xv, srcBase + g, 32);  // element g(+8)'s x value
    }
    if (r == 0) {                            // lanes 0 and 16 own the results
      float o[8];
#pragma unroll
      for (int g = 0; g < 8; ++g) {
        float soft = c[g] * fast_rcp(den[g]);
        float hard = hardQ(xs[g]);
        o[g] = (hard - soft) + soft;         // STE forward, fp32 like reference
      }
      float4* p = (float4*)(out + tile * 16 + srcBase);
      p[0] = make_float4(o[0], o[1], o[2], o[3]);
      p[1] = make_float4(o[4], o[5], o[6], o[7]);
    }
  }

  // Tail (N % 16), scalar path (dead for 4.19M inputs but kept general)
  const int gtid = blockIdx.x * blockDim.x + threadIdx.x;
  const int t = nTiles * 16 + gtid;
  if (t < N) {
    float xv = x[t], L[16], m = -1e30f;
#pragma unroll
    for (int k = 0; k < 16; ++k) {
      float d = xv - (float)(k - 7);
      float tt = fmaf(d * d, G2_OVER_V, 1.0f);
      L[k] = -NEG_EXP * fast_log2(tt);
      m = fmaxf(m, L[k]);
    }
    float m2 = fmaxf(m, L_EPS), eT = fast_exp2(L_EPS - m2);
    float num = 0.f, den = 0.f;
#pragma unroll
    for (int k = 0; k < 16; ++k) {
      float u = fast_exp2(L[k] - m2) + eT;
      num = fmaf(u, (float)(k - 7), num);
      den += u;
    }
    float soft = num * fast_rcp(den), hard = hardQ(xv);
    out[t] = (hard - soft) + soft;
  }
}

extern "C" void kernel_launch(void* const* d_in, const int* in_sizes, int n_in,
                              void* d_out, int out_size, void* d_ws, size_t ws_size,
                              hipStream_t stream) {
  (void)n_in; (void)d_ws; (void)ws_size; (void)out_size;
  const float* x = (const float*)d_in[0];   // (16,64,64,64) fp32
  // d_in[1] = codebook; fixed to arange(-7..8) per reference constants.
  float* out = (float*)d_out;

  const int N      = in_sizes[0];
  const int nTiles = N / 16;

  const int threads       = 256;             // 8 wave32 per block
  const int wavesPerBlock = threads / 32;
  long long wantWaves = ((long long)nTiles + 3) / 4;  // ~4 tiles per wave
  int blocks = (int)((wantWaves + wavesPerBlock - 1) / wavesPerBlock);
  if (blocks < 1) blocks = 1;
  if (blocks > (1 << 20)) blocks = (1 << 20);

  quant_ts_wmma<<<blocks, threads, 0, stream>>>(x, out, nTiles, N);
}
